// AttentionHead_68169720922224
// MI455X (gfx1250) — matrix operand order repaired and checked by reading, not compiled, not run on hardware
//
#include <hip/hip_runtime.h>
#include <hip/hip_bf16.h>

// ---- problem constants (match reference) ----
constexpr int NB = 8;     // batch
constexpr int NT = 2048;  // sequence length
constexpr int NF = 512;   // feature dim (F == C)
constexpr int ND = 64;    // head dim (D == O)
constexpr int TBLK = 16;  // WMMA tile edge
constexpr float SCALE = 0.125f; // 1/sqrt(64)

typedef __attribute__((ext_vector_type(16))) __bf16        v16bf;
typedef __attribute__((ext_vector_type(8)))  float         v8f;
typedef __attribute__((ext_vector_type(4)))  unsigned int  v4u;

// WMMA operand tile: slots s=2r+h per ISA 7.12.2.
//   A (16x32): lane row M=L&15; K = 16*(r>=4) + 8*(L>=16) + 2*(r&3)+h
//              -> u[0..7]  = K: half8+0..7   (contiguous)
//              -> u[8..15] = K: 16+half8+0..7 (contiguous)
//   B (32x16): lane col N=L&15; K same as A -> u[0..7], u[8..15] as above
union ABf {
  unsigned short u[16];
  __bf16         h[16];
  v4u            q[2];   // two 16-byte halves
  v16bf          v;
};
union H8 {
  unsigned short u[8];
  __bf16         h[8];
  v4u            q;
};

// bf16 tile loaders: p must point at the first element of the lane's run
__device__ __forceinline__ void load_a_bf16(ABf& a, const unsigned short* p) {
  a.q[0] = *(const v4u*)(p);        // K = base .. base+7
  a.q[1] = *(const v4u*)(p + 16);   // K = base+16 .. base+23
}
__device__ __forceinline__ void load_b_bf16(ABf& b, const unsigned short* p) {
  b.q[0] = *(const v4u*)(p);        // K = base .. base+7
  b.q[1] = *(const v4u*)(p + 16);   // K = base+16 .. base+23
}

// =====================================================================
// Kernel 1: q/k/v projections.  out(16 rows x 64 cols) = In(16x512) @ W^T + b
// q,k stored row-major [(b*T+t)*64 + d]; v stored TRANSPOSED [(b*64+o)*T + t]
// grid.x = NB*(NT/16)*3 ; block = 128 (4 waves, one 16-col d-chunk each)
// =====================================================================
__global__ void qkv_proj_kernel(const float* __restrict__ X, const float* __restrict__ Z,
                                const float* __restrict__ Wq, const float* __restrict__ bq,
                                const float* __restrict__ Wk, const float* __restrict__ bk,
                                const float* __restrict__ Wv, const float* __restrict__ bv,
                                unsigned short* __restrict__ qbf,
                                unsigned short* __restrict__ kbf,
                                unsigned short* __restrict__ vbfT) {
  const int lane = threadIdx.x & 31;
  const int wave = threadIdx.x >> 5;          // d-chunk: 0..3
  const int mat  = blockIdx.x % 3;            // 0=q, 1=k, 2=v(transposed)
  const int rb   = blockIdx.x / 3;            // row-block over NB*NT/16
  const int b    = rb / (NT / TBLK);
  const int tb   = rb % (NT / TBLK);

  const float* In   = (mat == 0) ? X  : Z;
  const float* W    = (mat == 0) ? Wq : ((mat == 1) ? Wk : Wv);
  const float* bias = (mat == 0) ? bq : ((mat == 1) ? bk : bv);

  const int d0 = wave * TBLK;
  const int col = lane & 15;
  const int half8  = (lane >= 16) ? 8 : 0;
  const int half16 = (lane >= 16) ? 16 : 0;
  const float* inRow = In + ((size_t)b * NT + (size_t)tb * TBLK) * NF;

  v8f acc = {0.f, 0.f, 0.f, 0.f, 0.f, 0.f, 0.f, 0.f};
  for (int kc = 0; kc < NF; kc += 32) {
    ABf a, w;
    const float* pa = inRow + (size_t)col * NF + kc + half16;
    const float* pw = W + (size_t)(d0 + col) * NF + kc + half16;
#pragma unroll
    for (int t = 0; t < 8; ++t) {
      a.h[t]     = (__bf16)pa[t];        // K = half16 + t
      a.h[8 + t] = (__bf16)pa[8 + t];    // K = half16 + 8 + t
    }
#pragma unroll
    for (int t = 0; t < 16; ++t) w.h[t] = (__bf16)pw[t];  // K = half16 + t
    acc = __builtin_amdgcn_wmma_f32_16x16x32_bf16(false, a.v, false, w.v,
                                                  (short)0, acc, false, false);
  }
  const float bia = bias[d0 + col];

  if (mat == 2) {
    // transposed store: lane owns column o = d0+col, 8 consecutive t's -> 1x b128
    H8 st;
#pragma unroll
    for (int vR = 0; vR < 8; ++vR) st.h[vR] = (__bf16)(acc[vR] + bia);
    *(v4u*)&vbfT[((size_t)b * ND + d0 + col) * NT + (size_t)tb * TBLK + half8] = st.q;
  } else {
    unsigned short* Out = (mat == 0) ? qbf : kbf;
#pragma unroll
    for (int vR = 0; vR < 8; ++vR) {
      const int row = tb * TBLK + vR + half8;
      union { __bf16 h; unsigned short us; } c; c.h = (__bf16)(acc[vR] + bia);
      Out[((size_t)b * NT + row) * ND + d0 + col] = c.us;
    }
  }
}

// =====================================================================
// Kernel 2: per-context-row softmax stats m[i], L[i] over j <= i.
// grid.x = NB*(NT/16) ; block = 32 (one wave per 16-row i-block)
// =====================================================================
__global__ void softmax_stats_kernel(const unsigned short* __restrict__ qbf,
                                     const unsigned short* __restrict__ kbf,
                                     float* __restrict__ Ms, float* __restrict__ Ls) {
  const int lane = threadIdx.x & 31;
  const int b  = blockIdx.x / (NT / TBLK);
  const int ib = blockIdx.x % (NT / TBLK);
  const int col = lane & 15;
  const int half8  = (lane >= 16) ? 8 : 0;
  const int half16 = (lane >= 16) ? 16 : 0;

  // Loop-invariant A tiles: k rows i, K-dim = head dim (0..31, 32..63)
  const unsigned short* krow = kbf + ((size_t)b * NT + (size_t)ib * TBLK) * ND;
  ABf a0, a1;
  load_a_bf16(a0, krow + (size_t)col * ND + half8);
  load_a_bf16(a1, krow + (size_t)col * ND + 32 + half8);

  float m_run[8], l_run[8];
#pragma unroll
  for (int vR = 0; vR < 8; ++vR) { m_run[vR] = -__builtin_inff(); l_run[vR] = 0.f; }

  for (int jb = 0; jb <= ib; ++jb) {
    const unsigned short* qrow = qbf + ((size_t)b * NT + (size_t)jb * TBLK) * ND;
    if (jb < ib) __builtin_prefetch(qrow + TBLK * ND + lane * 64, 0, 1);
    ABf b0, b1;
    load_b_bf16(b0, qrow + (size_t)col * ND + half8);
    load_b_bf16(b1, qrow + (size_t)col * ND + 32 + half8);

    v8f sc = {0.f, 0.f, 0.f, 0.f, 0.f, 0.f, 0.f, 0.f};
    sc = __builtin_amdgcn_wmma_f32_16x16x32_bf16(false, a0.v, false, b0.v, (short)0, sc, false, false);
    sc = __builtin_amdgcn_wmma_f32_16x16x32_bf16(false, a1.v, false, b1.v, (short)0, sc, false, false);

    const int j = jb * TBLK + col;
#pragma unroll
    for (int vR = 0; vR < 8; ++vR) {
      const int i = ib * TBLK + vR + half8;
      float s = sc[vR] * SCALE;
      if (j > i) s = -__builtin_inff();           // strictly-upper-triangle mask
      float rmax = s;
#pragma unroll
      for (int m = 1; m < 16; m <<= 1) rmax = fmaxf(rmax, __shfl_xor(rmax, m, 16));
      float rsum = __expf(s - rmax);
#pragma unroll
      for (int m = 1; m < 16; m <<= 1) rsum += __shfl_xor(rsum, m, 16);
      const float newm = fmaxf(m_run[vR], rmax);
      l_run[vR] = l_run[vR] * __expf(m_run[vR] - newm) + rsum * __expf(rmax - newm);
      m_run[vR] = newm;
    }
  }
  if (col == 0) {
#pragma unroll
    for (int vR = 0; vR < 8; ++vR) {
      const int i = ib * TBLK + vR + half8;
      Ms[(size_t)b * NT + i] = m_run[vR];
      Ls[(size_t)b * NT + i] = l_run[vR];
    }
  }
}

// =====================================================================
// Kernel 3: out[j,0:64] = sum_{i>=j} exp(s[i,j]-m[i])/L[i] * v[i,0:64]
// grid.x = NB*(NT/16) ; block = 32 (one wave per 16-query j-block)
// =====================================================================
__global__ void attn_out_kernel(const unsigned short* __restrict__ qbf,
                                const unsigned short* __restrict__ kbf,
                                const unsigned short* __restrict__ vbfT,
                                const float* __restrict__ Ms, const float* __restrict__ Ls,
                                float* __restrict__ Out) {
  // P^T tile in LDS: row = j_local, col = i_local (0..31); row padded to 40
  // ushorts (80B) -> 16B-aligned vector access + conflict-free bank stride.
  __shared__ unsigned short pbufT[16][40];
  const int lane = threadIdx.x & 31;
  const int b  = blockIdx.x / (NT / TBLK);
  const int jb = blockIdx.x % (NT / TBLK);
  const int col = lane & 15;
  const int half8  = (lane >= 16) ? 8 : 0;
  const int half16 = (lane >= 16) ? 16 : 0;
  const int j = jb * TBLK + col;

  // Loop-invariant B tiles: q cols j, K-dim = head dim
  const unsigned short* qrow = qbf + ((size_t)b * NT + (size_t)jb * TBLK) * ND;
  ABf bq0, bq1;
  load_b_bf16(bq0, qrow + (size_t)col * ND + half8);
  load_b_bf16(bq1, qrow + (size_t)col * ND + 32 + half8);

  v8f acc[4];
#pragma unroll
  for (int oc = 0; oc < 4; ++oc) acc[oc] = (v8f){0.f, 0.f, 0.f, 0.f, 0.f, 0.f, 0.f, 0.f};

  const int ii0 = ((jb >> 1) << 5);             // 32-aligned start; below-diag rows mask to 0
  for (int ii = ii0; ii < NT; ii += 32) {
    if (ii + 32 < NT)
      __builtin_prefetch(kbf + ((size_t)b * NT + ii + 32) * ND + lane * 128, 0, 1);
    // ---- build P (32 i-rows x 16 j-cols): two score WMMAs + softmax weights ----
#pragma unroll
    for (int sb = 0; sb < 2; ++sb) {
      const int ibase = ii + sb * TBLK;
      const unsigned short* krow = kbf + ((size_t)b * NT + ibase) * ND;
      ABf a0, a1;
      load_a_bf16(a0, krow + (size_t)col * ND + half8);
      load_a_bf16(a1, krow + (size_t)col * ND + 32 + half8);
      v8f sc = {0.f, 0.f, 0.f, 0.f, 0.f, 0.f, 0.f, 0.f};
      sc = __builtin_amdgcn_wmma_f32_16x16x32_bf16(false, a0.v, false, bq0.v, (short)0, sc, false, false);
      sc = __builtin_amdgcn_wmma_f32_16x16x32_bf16(false, a1.v, false, bq1.v, (short)0, sc, false, false);
      H8 ph;
#pragma unroll
      for (int vR = 0; vR < 8; ++vR) {
        const int i = ibase + vR + half8;
        float p = 0.f;
        if (j <= i) {
          const float mi = Ms[(size_t)b * NT + i];
          const float li = Ls[(size_t)b * NT + i];
          p = __expf(sc[vR] * SCALE - mi) / li;
        }
        ph.h[vR] = (__bf16)p;
      }
      // one 16-byte DS store: P^T[j_local][ibase_local + half8 .. +7]
      *(v4u*)&pbufT[col][sb * TBLK + half8] = ph.q;
    }
    __syncthreads();   // single-wave WG: ordering fence for LDS round-trip

    // ---- out(16j x 64) += P^T(16x32) @ V(32x64) ----
    ABf ap;   // A element (M=j_local, K=i_local) = pbufT[j_local][K]
    ap.q[0] = *(const v4u*)&pbufT[col][half8];        // K = half8 + 0..7
    ap.q[1] = *(const v4u*)&pbufT[col][16 + half8];   // K = 16 + half8 + 0..7
#pragma unroll
    for (int oc = 0; oc < 4; ++oc) {
      ABf bv;   // B element (K=i_local, N=o) = vT[(b*64+o)*T + ii + K]
      load_b_bf16(bv, vbfT + ((size_t)b * ND + oc * TBLK + col) * NT + ii + half8);
      acc[oc] = __builtin_amdgcn_wmma_f32_16x16x32_bf16(false, ap.v, false, bv.v,
                                                        (short)0, acc[oc], false, false);
    }
    __syncthreads();   // pbufT reused next iteration
  }

#pragma unroll
  for (int oc = 0; oc < 4; ++oc)
#pragma unroll
    for (int vR = 0; vR < 8; ++vR) {
      const int row = jb * TBLK + vR + half8;
      Out[((size_t)b * NT + row) * ND + oc * TBLK + col] = acc[oc][vR];
    }
}

// =====================================================================
extern "C" void kernel_launch(void* const* d_in, const int* in_sizes, int n_in,
                              void* d_out, int out_size, void* d_ws, size_t ws_size,
                              hipStream_t stream) {
  (void)in_sizes; (void)n_in; (void)out_size; (void)ws_size;
  const float* X  = (const float*)d_in[0];
  const float* Z  = (const float*)d_in[1];
  const float* Wq = (const float*)d_in[2];
  const float* bq = (const float*)d_in[3];
  const float* Wk = (const float*)d_in[4];
  const float* bk = (const float*)d_in[5];
  const float* Wv = (const float*)d_in[6];
  const float* bv = (const float*)d_in[7];
  float* Out = (float*)d_out;

  // workspace layout: bf16 q/k (row-major) + v (transposed) + fp32 m/L
  unsigned short* qbf  = (unsigned short*)d_ws;
  unsigned short* kbf  = qbf + (size_t)NB * NT * ND;
  unsigned short* vbfT = kbf + (size_t)NB * NT * ND;
  float* Ms = (float*)(vbfT + (size_t)NB * NT * ND);
  float* Ls = Ms + (size_t)NB * NT;

  const int rowBlocks = NB * (NT / TBLK);   // 1024

  qkv_proj_kernel<<<rowBlocks * 3, 128, 0, stream>>>(X, Z, Wq, bq, Wk, bk, Wv, bv,
                                                     qbf, kbf, vbfT);
  softmax_stats_kernel<<<rowBlocks, 32, 0, stream>>>(qbf, kbf, Ms, Ls);
  attn_out_kernel<<<rowBlocks, 32, 0, stream>>>(qbf, kbf, vbfT, Ms, Ls, Out);
}
